// CompressedCausalAttention_17575006175526
// MI455X (gfx1250) — compile-verified, matrix operand order checked by reading, not run on hardware
//
#include <hip/hip_runtime.h>
#include <hip/hip_bf16.h>

#define S_  384
#define B_  8
#define C_  1024
#define H_  16
#define M_  1536
#define CC_ 64
#define T_  (M_ + S_)   // 1920
#define ROWS_ (S_ * B_) // 3072

typedef __bf16 bf16_t;
typedef bf16_t bf16x16 __attribute__((ext_vector_type(16)));
typedef float  f32x8   __attribute__((ext_vector_type(8)));

// ---------------------------------------------------------------------------
// WMMA helpers
// ---------------------------------------------------------------------------
__device__ __forceinline__ f32x8 wmma_bf16(bf16x16 a, bf16x16 b, f32x8 c) {
  return __builtin_amdgcn_wmma_f32_16x16x32_bf16(false, a, false, b, (short)0, c,
                                                 false, false);
}

// Load one 16x32 operand (A, or B^T rows) from row-major [rows][ld] bf16.
// ISA 16-bit A layout: lane l (group g = lane>>4) holds row (row0 + (lane&15)),
// K offsets {k0+8g .. +8} and {k0+16+8g .. +8}.
__device__ __forceinline__ bf16x16 load_frag(const bf16_t* __restrict__ base,
                                             int row0, int ld, int k0, int lane) {
  const int g = lane >> 4, l = lane & 15;
  const bf16_t* p = base + (size_t)(row0 + l) * ld + k0 + g * 8;
  bf16x16 r;
  uint4* rp = reinterpret_cast<uint4*>(&r);
  rp[0] = *reinterpret_cast<const uint4*>(p);
  rp[1] = *reinterpret_cast<const uint4*>(p + 16);
  return r;
}

// ---------------------------------------------------------------------------
// Prep kernels
// ---------------------------------------------------------------------------
__global__ void k_prep_xp(const float* __restrict__ x, const float* __restrict__ pe,
                          bf16_t* __restrict__ xp, int n) {
  int i = blockIdx.x * blockDim.x + threadIdx.x;
  if (i < n) xp[i] = (bf16_t)(x[i] + pe[i]);
}

// w [rows][cols] f32  ->  wT [cols][rows] bf16  (B^T rows contiguous over K)
__global__ void k_transpose_bf(const float* __restrict__ w, bf16_t* __restrict__ wT,
                               int rows, int cols) {
  int i = blockIdx.x * blockDim.x + threadIdx.x;
  if (i < rows * cols) {
    int r = i / cols, c = i % cols;
    wT[(size_t)c * rows + r] = (bf16_t)w[i];
  }
}

// memory_k/v [M][B][H][CC] f32 -> k_bf [B*H][T][CC], vT_bf [B*H][CC][T] (bf16)
__global__ void k_prep_mem(const float* __restrict__ mk, const float* __restrict__ mv,
                           bf16_t* __restrict__ k_bf, bf16_t* __restrict__ vT_bf,
                           int n) {
  int i = blockIdx.x * blockDim.x + threadIdx.x;
  if (i < n) {
    int cc = i % CC_;
    int r  = i / CC_;
    int h  = r % H_; r /= H_;
    int b  = r % B_;
    int t  = r / B_;
    size_t bh = (size_t)(b * H_ + h);
    k_bf [(bh * T_  + t ) * CC_ + cc] = (bf16_t)mk[i];
    vT_bf[(bh * CC_ + cc) * T_  + t ] = (bf16_t)mv[i];
  }
}

// ---------------------------------------------------------------------------
// 64x64-per-wave GEMM core: C[64x64] += A[64xK] * BT[64xK]^T
// 16 accumulators; each B fragment is reused by 4 A fragments.
// ---------------------------------------------------------------------------
__device__ __forceinline__ void gemm_core64(const bf16_t* __restrict__ A, int lda,
                                            const bf16_t* __restrict__ BT, int ldb,
                                            int row0, int col0, int K, int lane,
                                            f32x8 c[16]) {
  for (int k0 = 0; k0 < K; k0 += 32) {
    bf16x16 a[4];
#pragma unroll
    for (int i = 0; i < 4; ++i) a[i] = load_frag(A, row0 + 16 * i, lda, k0, lane);
#pragma unroll
    for (int j = 0; j < 4; ++j) {
      bf16x16 bj = load_frag(BT, col0 + 16 * j, ldb, k0, lane);
#pragma unroll
      for (int i = 0; i < 4; ++i) c[i * 4 + j] = wmma_bf16(a[i], bj, c[i * 4 + j]);
    }
  }
}

// GEMM1: qkv = xp @ w_qkv + b_qkv, scattered into q/k/vT buffers
__global__ __launch_bounds__(128) void k_gemm_qkv(
    const bf16_t* __restrict__ A, const bf16_t* __restrict__ BT,
    const float* __restrict__ bias,
    bf16_t* __restrict__ q_bf, bf16_t* __restrict__ k_bf,
    bf16_t* __restrict__ vT_bf) {
  const int lane = threadIdx.x & 31;
  const int wave = threadIdx.x >> 5;
  const int g = lane >> 4, n = lane & 15;
  const int col0 = blockIdx.x * 64;
  const int row0 = blockIdx.y * 256 + wave * 64;

  f32x8 c[16] = {};
  gemm_core64(A, C_, BT, C_, row0, col0, C_, lane, c);

  const int region = col0 >> 10;  // 0=q 1=k 2=v (64-col strip never crosses)
#pragma unroll
  for (int i = 0; i < 4; ++i)
#pragma unroll
    for (int j = 0; j < 4; ++j) {
      const int col  = col0 + 16 * j + n;
      const int cidx = col & (C_ - 1);
      const int h = cidx >> 6, cc = cidx & 63;
      const float bv = bias[col];
#pragma unroll
      for (int r = 0; r < 8; ++r) {
        const int row = row0 + 16 * i + 8 * g + r;
        const int s = row >> 3, b = row & 7;  // row = s*B + b
        const float v = c[i * 4 + j][r] + bv;
        const size_t bh = (size_t)(b * H_ + h);
        if (region == 0)      q_bf [(bh * S_  + s        ) * CC_ + cc] = (bf16_t)v;
        else if (region == 1) k_bf [(bh * T_  + (M_ + s) ) * CC_ + cc] = (bf16_t)v;
        else                  vT_bf[(bh * CC_ + cc) * T_ + (M_ + s)]   = (bf16_t)v;
      }
    }
}

// GEMM2: out = attn @ w_out + b_out (f32 output)
__global__ __launch_bounds__(128) void k_gemm_out(
    const bf16_t* __restrict__ A, const bf16_t* __restrict__ BT,
    const float* __restrict__ bias, float* __restrict__ out) {
  const int lane = threadIdx.x & 31;
  const int wave = threadIdx.x >> 5;
  const int g = lane >> 4, n = lane & 15;
  const int col0 = blockIdx.x * 64;
  const int row0 = blockIdx.y * 256 + wave * 64;

  f32x8 c[16] = {};
  gemm_core64(A, C_, BT, C_, row0, col0, C_, lane, c);

#pragma unroll
  for (int i = 0; i < 4; ++i)
#pragma unroll
    for (int j = 0; j < 4; ++j) {
      const int col = col0 + 16 * j + n;
      const float bv = bias[col];
#pragma unroll
      for (int r = 0; r < 8; ++r) {
        const int row = row0 + 16 * i + 8 * g + r;
        out[(size_t)row * C_ + col] = c[i * 4 + j][r] + bv;
      }
    }
}

// ---------------------------------------------------------------------------
// Flash attention: one wave per (b,h, 32-query strip = two 16-row tiles),
// 32-key chunks; K/V fragments loaded once and shared by both query tiles.
// ---------------------------------------------------------------------------
__global__ __launch_bounds__(32) void k_attn(
    const bf16_t* __restrict__ q_bf, const bf16_t* __restrict__ k_bf,
    const bf16_t* __restrict__ vT_bf,
    const unsigned char* __restrict__ terminal,
    const unsigned char* __restrict__ content_mask,   // [S][S][B], 1 = masked
    const unsigned char* __restrict__ padding_mask,   // [S][B],   1 = masked
    const int* __restrict__ memory_length,
    bf16_t* __restrict__ attn_bf) {                   // [S*B][C]
  __shared__ bf16_t Plds[16][40];  // 16x32 P tile, padded rows (80B, 16B-aligned)

  const int lane = threadIdx.x & 31;
  const int g = lane >> 4, n = lane & 15;
  const int qt = blockIdx.x % (S_ / 32);
  const int bh = blockIdx.x / (S_ / 32);
  const int b = bh / H_, h = bh % H_;
  const int q0 = qt * 32;

  const bf16_t* Q  = q_bf  + (size_t)bh * S_ * CC_;
  const bf16_t* K  = k_bf  + (size_t)bh * T_ * CC_;
  const bf16_t* VT = vT_bf + (size_t)bh * CC_ * T_;

  bf16x16 aq[2][2];
#pragma unroll
  for (int t = 0; t < 2; ++t) {
    aq[t][0] = load_frag(Q, q0 + 16 * t, CC_, 0,  lane);
    aq[t][1] = load_frag(Q, q0 + 16 * t, CC_, 32, lane);
  }

  const int memlen = terminal[b] ? 0 : memory_length[b];

  float m_run[2][8], l_run[2][8];
  f32x8 o[2][4] = {};
#pragma unroll
  for (int t = 0; t < 2; ++t)
#pragma unroll
    for (int r = 0; r < 8; ++r) { m_run[t][r] = -1e30f; l_run[t][r] = 0.f; }

  auto chunk = [&](int t0, bool token_region) {
    // ---- K and V fragments for keys [t0, t0+32), shared by both q tiles ----
    bf16x16 bk[4];
    bk[0] = load_frag(K, t0,      CC_, 0,  lane);
    bk[1] = load_frag(K, t0,      CC_, 32, lane);
    bk[2] = load_frag(K, t0 + 16, CC_, 0,  lane);
    bk[3] = load_frag(K, t0 + 16, CC_, 32, lane);
    bf16x16 bv[4];
#pragma unroll
    for (int j = 0; j < 4; ++j) bv[j] = load_frag(VT, 16 * j, T_, t0, lane);

    const float scale = 0.125f;  // 1/sqrt(CC)
    // column-wise mask terms (independent of query row)
    const bool mm0 = (t0 + n) >= memlen;
    const bool mm1 = (t0 + 16 + n) >= memlen;
    const int tt0 = t0 - M_ + n;
    const int tt1 = tt0 + 16;
    const bool pad0 = token_region ? (padding_mask[tt0 * B_ + b] != 0) : false;
    const bool pad1 = token_region ? (padding_mask[tt1 * B_ + b] != 0) : false;

#pragma unroll
    for (int t = 0; t < 2; ++t) {
      // ---- scores: two 16x16 tiles, contract CC=64 ----
      f32x8 s0{}, s1{};
      s0 = wmma_bf16(aq[t][0], bk[0], s0);
      s0 = wmma_bf16(aq[t][1], bk[1], s0);
      s1 = wmma_bf16(aq[t][0], bk[2], s1);
      s1 = wmma_bf16(aq[t][1], bk[3], s1);

      float p0[8], p1[8];
      if (!token_region) {
#pragma unroll
        for (int r = 0; r < 8; ++r) {
          p0[r] = mm0 ? -1e9f : s0[r] * scale;
          p1[r] = mm1 ? -1e9f : s1[r] * scale;
        }
      } else {
#pragma unroll
        for (int r = 0; r < 8; ++r) {
          const int srow = q0 + 16 * t + 8 * g + r;
          const bool c0 = pad0 || (content_mask[((size_t)srow * S_ + tt0) * B_ + b] != 0);
          const bool c1 = pad1 || (content_mask[((size_t)srow * S_ + tt1) * B_ + b] != 0);
          p0[r] = c0 ? -1e9f : s0[r] * scale;
          p1[r] = c1 ? -1e9f : s1[r] * scale;
        }
      }
      // ---- online softmax (rows live across the 16 lanes of each half) ----
      float cm[8];
#pragma unroll
      for (int r = 0; r < 8; ++r) cm[r] = fmaxf(p0[r], p1[r]);
#pragma unroll
      for (int off = 1; off < 16; off <<= 1)
#pragma unroll
        for (int r = 0; r < 8; ++r) cm[r] = fmaxf(cm[r], __shfl_xor(cm[r], off, 32));
      float corr[8], rs[8];
#pragma unroll
      for (int r = 0; r < 8; ++r) {
        const float mnew = fmaxf(m_run[t][r], cm[r]);
        corr[r] = __expf(m_run[t][r] - mnew);
        m_run[t][r] = mnew;
        p0[r] = __expf(p0[r] - mnew);
        p1[r] = __expf(p1[r] - mnew);
        rs[r] = p0[r] + p1[r];
      }
#pragma unroll
      for (int off = 1; off < 16; off <<= 1)
#pragma unroll
        for (int r = 0; r < 8; ++r) rs[r] += __shfl_xor(rs[r], off, 32);
#pragma unroll
      for (int r = 0; r < 8; ++r) l_run[t][r] = l_run[t][r] * corr[r] + rs[r];
#pragma unroll
      for (int j = 0; j < 4; ++j)
#pragma unroll
        for (int r = 0; r < 8; ++r) o[t][j][r] *= corr[r];
      // ---- reshape P (C-layout) -> A-operand layout via LDS ----
      __syncthreads();
#pragma unroll
      for (int r = 0; r < 8; ++r) {
        Plds[8 * g + r][n]      = (bf16_t)p0[r];
        Plds[8 * g + r][16 + n] = (bf16_t)p1[r];
      }
      __syncthreads();
      bf16x16 ap;
      {
        uint4* app = reinterpret_cast<uint4*>(&ap);
        const bf16_t* prow = &Plds[n][g * 8];
        app[0] = *reinterpret_cast<const uint4*>(prow);
        app[1] = *reinterpret_cast<const uint4*>(prow + 16);
      }
      // ---- O += P @ V ----
#pragma unroll
      for (int j = 0; j < 4; ++j) o[t][j] = wmma_bf16(ap, bv[j], o[t][j]);
    }
  };

  for (int t0 = 0; t0 < memlen; t0 += 32) chunk(t0, false);
  for (int t0 = M_; t0 < M_ + S_; t0 += 32) {
    if (t0 - M_ > q0 + 31) break;  // fully-masked causal chunks for both tiles
    chunk(t0, true);
  }

#pragma unroll
  for (int t = 0; t < 2; ++t) {
    float inv[8];
#pragma unroll
    for (int r = 0; r < 8; ++r) inv[r] = 1.0f / l_run[t][r];
#pragma unroll
    for (int j = 0; j < 4; ++j)
#pragma unroll
      for (int r = 0; r < 8; ++r) {
        const int srow = q0 + 16 * t + 8 * g + r;
        const float v = o[t][j][r] * inv[r];
        attn_bf[((size_t)srow * B_ + b) * C_ + h * CC_ + 16 * j + n] = (bf16_t)v;
      }
  }
}

// ---------------------------------------------------------------------------
// Launch
// ---------------------------------------------------------------------------
extern "C" void kernel_launch(void* const* d_in, const int* in_sizes, int n_in,
                              void* d_out, int out_size, void* d_ws, size_t ws_size,
                              hipStream_t stream) {
  (void)in_sizes; (void)n_in; (void)out_size; (void)ws_size;
  const float* x      = (const float*)d_in[0];
  const float* pe     = (const float*)d_in[1];
  const float* mk     = (const float*)d_in[2];
  const float* mv     = (const float*)d_in[3];
  const float* w_qkv  = (const float*)d_in[4];
  const float* b_qkv  = (const float*)d_in[5];
  const float* w_out  = (const float*)d_in[6];
  const float* b_out  = (const float*)d_in[7];
  const unsigned char* terminal = (const unsigned char*)d_in[8];
  const unsigned char* cmask    = (const unsigned char*)d_in[9];
  const unsigned char* pmask    = (const unsigned char*)d_in[10];
  const int* memlen             = (const int*)d_in[11];
  float* out = (float*)d_out;

  char* ws = (char*)d_ws;
  bf16_t* xp    = (bf16_t*)ws; ws += (size_t)ROWS_ * C_ * 2;
  bf16_t* wqkvT = (bf16_t*)ws; ws += (size_t)3 * C_ * C_ * 2;
  bf16_t* woutT = (bf16_t*)ws; ws += (size_t)C_ * C_ * 2;
  bf16_t* qb    = (bf16_t*)ws; ws += (size_t)B_ * H_ * S_ * CC_ * 2;
  bf16_t* kb    = (bf16_t*)ws; ws += (size_t)B_ * H_ * T_ * CC_ * 2;
  bf16_t* vTb   = (bf16_t*)ws; ws += (size_t)B_ * H_ * T_ * CC_ * 2;
  bf16_t* attn  = (bf16_t*)ws; ws += (size_t)ROWS_ * C_ * 2;

  const int n_xp  = ROWS_ * C_;          // 3,145,728
  const int n_wq  = C_ * 3 * C_;         // 3,145,728
  const int n_wo  = C_ * C_;             // 1,048,576
  const int n_mem = M_ * B_ * H_ * CC_;  // 12,582,912

  k_prep_xp<<<(n_xp + 255) / 256, 256, 0, stream>>>(x, pe, xp, n_xp);
  k_transpose_bf<<<(n_wq + 255) / 256, 256, 0, stream>>>(w_qkv, wqkvT, C_, 3 * C_);
  k_transpose_bf<<<(n_wo + 255) / 256, 256, 0, stream>>>(w_out, woutT, C_, C_);
  k_prep_mem<<<(n_mem + 255) / 256, 256, 0, stream>>>(mk, mv, kb, vTb, n_mem);

  k_gemm_qkv<<<dim3(3 * C_ / 64, ROWS_ / 256), 128, 0, stream>>>(
      xp, wqkvT, b_qkv, qb, kb, vTb);
  k_attn<<<B_ * H_ * (S_ / 32), 32, 0, stream>>>(
      qb, kb, vTb, terminal, cmask, pmask, memlen, attn);
  k_gemm_out<<<dim3(C_ / 64, ROWS_ / 256), 128, 0, stream>>>(
      attn, woutT, b_out, out);
}
